// MultiheadAttention_24111946400368
// MI455X (gfx1250) — matrix operand order repaired and checked by reading, not compiled, not run on hardware
//
#include <hip/hip_runtime.h>

// Problem constants (from reference): T=2048, B=2, E=1024, H=16, D=64
#define TT 2048
#define BB 2
#define EE 1024
#define HH 16
#define DD 64
#define MROWS (TT * BB)   // 4096 rows of x when [T,B,E] is viewed as [T*B, E]
#define N_QKV (3 * EE)    // 3072
static constexpr float SCALING = 0.125f;  // D^-0.5

typedef __attribute__((ext_vector_type(16))) __bf16 v16bf;
typedef __attribute__((ext_vector_type(8)))  float  v8f;
typedef __attribute__((ext_vector_type(4)))  unsigned int u32x4;
typedef __attribute__((ext_vector_type(8)))  unsigned int u32x8;

struct U8 { unsigned int u[8]; };

__device__ __forceinline__ v8f zero8() {
  v8f z;
#pragma unroll
  for (int i = 0; i < 8; ++i) z[i] = 0.0f;
  return z;
}

__device__ __forceinline__ unsigned short f32_bf16(float x) {
  unsigned int u = __builtin_bit_cast(unsigned int, x);
  u += 0x7fffu + ((u >> 16) & 1u);          // round-to-nearest-even
  return (unsigned short)(u >> 16);
}

__device__ __forceinline__ unsigned int pack_bf16(float a, float b) {
  return (unsigned int)f32_bf16(a) | ((unsigned int)f32_bf16(b) << 16);
}

// Truncate a generic pointer to a shared variable to its LDS byte offset
// (flat->LDS aperture keeps the LDS offset in addr[31:0]).
__device__ __forceinline__ unsigned int lds_off(const void* p) {
  return (unsigned int)(unsigned long long)p;
}

// Tensor Data Mover: 2-D tile load Global -> LDS (bf16 elements, data_size=2B).
// Builds D# group0/group1 per CDNA5 ISA and issues the
// 2-SGPR-group form of tensor_load_to_lds. Call from ONE wave per workgroup;
// completion tracked with TENSORcnt (s_wait_tensorcnt 0 before the barrier).
__device__ __forceinline__ void tdm_load_2d_bf16(
    unsigned int lds_byte_off, const void* gaddr,
    unsigned int tile_w, unsigned int tile_h,
    unsigned int row_stride_elems, unsigned int tensor_w, unsigned int tensor_h) {
  unsigned long long ga = (unsigned long long)gaddr;
  u32x4 g0;
  g0[0] = 1u;                          // count=1, user descriptor, no gather
  g0[1] = lds_byte_off;                // D#.lds_addr (bytes)
  g0[2] = (unsigned int)ga;            // global_addr[31:0]
  g0[3] = ((unsigned int)(ga >> 32) & 0x01ffffffu) | 0x80000000u;  // addr[56:32] | type=2
  u32x8 g1;
  g1[0] = 0x00010000u;                 // workgroup_mask=0, data_size=1 (2 bytes)
  g1[1] = (tensor_w & 0xffffu) << 16;  // tensor_dim0[15:0]
  g1[2] = ((tensor_w >> 16) & 0xffffu) | ((tensor_h & 0xffffu) << 16);
  g1[3] = ((tensor_h >> 16) & 0xffffu) | ((tile_w & 0xffffu) << 16);  // tile_dim0
  g1[4] = tile_h & 0xffffu;            // tile_dim1 (tile_dim2 = 0 -> 2-D)
  g1[5] = row_stride_elems;            // tensor_dim0_stride[31:0]
  g1[6] = 0u;
  g1[7] = 0u;
  asm volatile("tensor_load_to_lds %0, %1" :: "s"(g0), "s"(g1) : "memory");
}

// A-matrix 16x32 bf16 fragment (ISA 7.12.2): lane holds row M=lane&15.
// uint j (packed K pair): j<4 -> K = half*8 + 2j ; j>=4 -> K = 16 + half*8 + 2(j-4).
// f selects which 32-wide K slice of a 64-wide row (row stored as K-pairs, 32 uints).
__device__ __forceinline__ v16bf load_a_frag(const unsigned int* row, int half, int f) {
  U8 t;
#pragma unroll
  for (int j = 0; j < 8; ++j) {
    int idx = f * 16 + ((j < 4) ? (half * 4 + j) : (8 + half * 4 + (j - 4)));
    t.u[j] = row[idx];
  }
  return __builtin_bit_cast(v16bf, t);
}

// B-matrix 32x16 bf16 fragment: lane holds column N=lane&15; uint j: j<4 -> K = half*8 + 2j ; j>=4 -> K = 16 + half*8 + 2(j-4).
__device__ __forceinline__ v16bf load_b_frag(const unsigned int* row, int half, int f) {
  U8 t;
#pragma unroll
  for (int j = 0; j < 8; ++j) t.u[j] = row[f * 16 + ((j < 4) ? (half * 4 + j) : (8 + half * 4 + (j - 4)))];
  return __builtin_bit_cast(v16bf, t);
}

__device__ __forceinline__ v8f wmma_bf16(v16bf a, v16bf b, v8f c) {
  return __builtin_amdgcn_wmma_f32_16x16x32_bf16(false, a, false, b, (short)0, c,
                                                 false, false);
}

// ---------------------------------------------------------------------------
// Kernel 1: fused QKV projection. qkv = x @ W_in^T + b; q *= scaling.
// Output scattered into bf16 [B,H,T,D] buffers for Q,K,V.
// Block tile: 128 (M) x 64 (N), 8 waves, each wave owns a 16-row strip.
// (A/B tiles need f32->bf16 conversion in flight, so staged through VGPRs.)
// ---------------------------------------------------------------------------
__global__ __launch_bounds__(256) void qkv_proj_kernel(
    const float* __restrict__ x, const float* __restrict__ w,
    const float* __restrict__ bias, unsigned short* __restrict__ Qb,
    unsigned short* __restrict__ Kb, unsigned short* __restrict__ Vb) {
  __shared__ __align__(16) unsigned int As[128][16];  // 128 rows x 32 K (bf16 pairs)
  __shared__ __align__(16) unsigned int Bs[64][16];   // 64 N-rows x 32 K

  const int tid = threadIdx.x, lane = tid & 31, wave = tid >> 5;
  const int half = lane >> 4, ln = lane & 15;
  const int m0 = blockIdx.y * 128, n0 = blockIdx.x * 64;

  v8f acc[4];
#pragma unroll
  for (int i = 0; i < 4; ++i) acc[i] = zero8();

  for (int k0 = 0; k0 < EE; k0 += 32) {
#pragma unroll
    for (int i = 0; i < 8; ++i) {  // A tile: 2048 uints / 256 thr
      int idx = tid + i * 256, r = idx >> 4, c = idx & 15;
      const float* p = x + (size_t)(m0 + r) * EE + k0 + c * 2;
      As[r][c] = pack_bf16(p[0], p[1]);
    }
#pragma unroll
    for (int i = 0; i < 4; ++i) {  // W tile: 1024 uints
      int idx = tid + i * 256, r = idx >> 4, c = idx & 15;
      const float* p = w + (size_t)(n0 + r) * EE + k0 + c * 2;
      Bs[r][c] = pack_bf16(p[0], p[1]);
    }
    __syncthreads();

    v16bf a = load_a_frag(&As[wave * 16 + ln][0], half, 0);
#pragma unroll
    for (int nt = 0; nt < 4; ++nt) {
      v16bf b = load_b_frag(&Bs[nt * 16 + ln][0], half, 0);
      acc[nt] = wmma_bf16(a, b, acc[nt]);
    }
    __syncthreads();
  }

  // Epilogue: C layout m = r + 8*half, n = lane&15 within each 16x16 tile.
#pragma unroll
  for (int nt = 0; nt < 4; ++nt) {
    int n = n0 + nt * 16 + ln;
    int which = n >> 10, e = n & 1023, hh = e >> 6, d = e & 63;
    unsigned short* dst = (which == 0) ? Qb : (which == 1 ? Kb : Vb);
    float bv = bias[n];
#pragma unroll
    for (int r = 0; r < 8; ++r) {
      int m = m0 + wave * 16 + r + 8 * half;
      int t = m >> 1, bidx = m & 1;  // row = t*B + b, B==2
      float v = acc[nt][r] + bv;
      if (which == 0) v *= SCALING;
      dst[(((size_t)(bidx * HH + hh) * TT + t) * DD) + d] = f32_bf16(v);
    }
  }
}

// ---------------------------------------------------------------------------
// Kernel 2: flash attention per (b, h, 128-row q-tile) with per-head bias and
// analytic causal mask. K blocks DMA'd into LDS by the Tensor Data Mover
// (wave 0 issues the descriptor, TENSORcnt gates the barrier); V blocks are
// transposed in flight so PV B-fragments read contiguous K-pairs.
// ---------------------------------------------------------------------------
__global__ __launch_bounds__(256) void attn_kernel(
    const unsigned short* __restrict__ Qb, const unsigned short* __restrict__ Kb,
    const unsigned short* __restrict__ Vb, const float* __restrict__ attn_bias,
    unsigned short* __restrict__ Ctx) {
  __shared__ __align__(16) unsigned int Ks[64][32];       // K block [s][d pairs] (TDM dest)
  __shared__ __align__(16) unsigned short Vt[64][64];     // V block transposed [d][s]
  __shared__ __align__(16) unsigned short Ps[8][16][64];  // per-wave P scratch

  const int tid = threadIdx.x, lane = tid & 31, wave = tid >> 5;
  const int half = lane >> 4, ln = lane & 15;
  const int bq = blockIdx.x, h = blockIdx.y, bb = blockIdx.z;
  const size_t head_off = (size_t)(bb * HH + h) * TT;
  const unsigned int ks_lds = lds_off(&Ks[0][0]);

  // Q fragments cached in registers for the whole kernel (16 rows x 64 K).
  const int qrow = bq * 128 + wave * 16 + ln;
  const unsigned int* qptr = (const unsigned int*)Qb + (head_off + qrow) * 32;
  v16bf qf0 = load_a_frag(qptr, half, 0);
  v16bf qf1 = load_a_frag(qptr, half, 1);

  v8f O[4];
#pragma unroll
  for (int i = 0; i < 4; ++i) O[i] = zero8();
  float mrow[8], lrow[8];
#pragma unroll
  for (int r = 0; r < 8; ++r) { mrow[r] = -1e30f; lrow[r] = 0.0f; }

  const float* biasp = attn_bias + (size_t)h * TT * TT;
  const int wmax = bq * 128 + wave * 16 + 15;  // last unmasked col for this wave
  const int smax = bq * 128 + 127;             // block-uniform loop bound

  for (int s0 = 0; s0 <= smax; s0 += 64) {
    // K tile 64x64 bf16: straight byte copy -> Tensor Data Mover, one wave.
    if (wave == 0) {
      tdm_load_2d_bf16(ks_lds, Kb + (head_off + s0) * DD,
                       /*tile_w=*/DD, /*tile_h=*/64,
                       /*row_stride=*/DD, /*tensor_w=*/DD, /*tensor_h=*/TT);
    }
    // V tile, transposed into [d][s] (needs element shuffle -> manual copy).
    const unsigned int* Vg = (const unsigned int*)Vb + (head_off + s0) * 32;
#pragma unroll
    for (int i = 0; i < 8; ++i) {
      int idx = tid + i * 256, s = idx >> 5, dp = idx & 31;
      unsigned int v = Vg[s * 32 + dp];
      Vt[2 * dp][s] = (unsigned short)(v & 0xffffu);
      Vt[2 * dp + 1][s] = (unsigned short)(v >> 16);
    }
    __builtin_amdgcn_s_wait_tensorcnt(0);  // wave 0: DMA done; others: no-op
    __syncthreads();

    if (s0 <= wmax) {  // wave-uniform: EXEC stays all-ones around WMMA
      v8f S[4];
#pragma unroll
      for (int i = 0; i < 4; ++i) S[i] = zero8();
#pragma unroll
      for (int f = 0; f < 2; ++f) {  // GEMM K-dim = D = 64 -> two 32-steps
        v16bf a = (f == 0) ? qf0 : qf1;
#pragma unroll
        for (int nt = 0; nt < 4; ++nt) {
          v16bf b = load_b_frag(&Ks[nt * 16 + ln][0], half, f);
          S[nt] = wmma_bf16(a, b, S[nt]);
        }
      }
      // scores += bias*scaling; apply causal mask analytically
#pragma unroll
      for (int nt = 0; nt < 4; ++nt) {
        int scol = s0 + nt * 16 + ln;
#pragma unroll
        for (int r = 0; r < 8; ++r) {
          int trow = bq * 128 + wave * 16 + r + 8 * half;
          float sv = S[nt][r] + biasp[(size_t)trow * TT + scol] * SCALING;
          S[nt][r] = (scol > trow) ? -1e9f : sv;
        }
      }
      // online softmax (row reductions across the 16 lanes of each half)
#pragma unroll
      for (int r = 0; r < 8; ++r) {
        float mx = fmaxf(fmaxf(S[0][r], S[1][r]), fmaxf(S[2][r], S[3][r]));
#pragma unroll
        for (int off = 1; off < 16; off <<= 1)
          mx = fmaxf(mx, __shfl_xor(mx, off, 32));
        float mi = fmaxf(mrow[r], mx);
        float alpha = __expf(mrow[r] - mi);
        mrow[r] = mi;
        float rs = 0.0f;
#pragma unroll
        for (int nt = 0; nt < 4; ++nt) {
          float p = __expf(S[nt][r] - mi);
          S[nt][r] = p;
          rs += p;
        }
#pragma unroll
        for (int off = 1; off < 16; off <<= 1) rs += __shfl_xor(rs, off, 32);
        lrow[r] = lrow[r] * alpha + rs;
#pragma unroll
        for (int dt = 0; dt < 4; ++dt) O[dt][r] *= alpha;
      }
      // P (C layout) -> per-wave LDS -> A layout; in-wave DS ops are in-order
#pragma unroll
      for (int nt = 0; nt < 4; ++nt)
#pragma unroll
        for (int r = 0; r < 8; ++r)
          Ps[wave][r + 8 * half][nt * 16 + ln] = f32_bf16(S[nt][r]);

      const unsigned int* prow = (const unsigned int*)&Ps[wave][ln][0];
      v16bf pf0 = load_a_frag(prow, half, 0);
      v16bf pf1 = load_a_frag(prow, half, 1);
#pragma unroll
      for (int f = 0; f < 2; ++f) {  // O += P x V, GEMM K-dim = 64 (s)
        v16bf a = (f == 0) ? pf0 : pf1;
#pragma unroll
        for (int dt = 0; dt < 4; ++dt) {
          v16bf b = load_b_frag((const unsigned int*)&Vt[dt * 16 + ln][0], half, f);
          O[dt] = wmma_bf16(a, b, O[dt]);
        }
      }
    }
    __syncthreads();  // no wave reads LDS when the next TDM is issued
  }

  // ctx back to [T*B, E] layout (bf16) for the output projection
#pragma unroll
  for (int dt = 0; dt < 4; ++dt) {
    int d = dt * 16 + ln;
#pragma unroll
    for (int r = 0; r < 8; ++r) {
      int trow = bq * 128 + wave * 16 + r + 8 * half;
      float o = O[dt][r] / lrow[r];
      Ctx[((size_t)trow * BB + bb) * EE + h * DD + d] = f32_bf16(o);
    }
  }
}

// ---------------------------------------------------------------------------
// Kernel 3: out = ctx @ W_out^T + b_out (fp32 output). The bf16 A-tile is a
// straight byte copy -> staged by the Tensor Data Mover each K-step.
// ---------------------------------------------------------------------------
__global__ __launch_bounds__(256) void out_proj_kernel(
    const unsigned short* __restrict__ CtxH, const float* __restrict__ w,
    const float* __restrict__ bias, float* __restrict__ out) {
  __shared__ __align__(16) unsigned int As[128][16];
  __shared__ __align__(16) unsigned int Bs[64][16];

  const int tid = threadIdx.x, lane = tid & 31, wave = tid >> 5;
  const int half = lane >> 4, ln = lane & 15;
  const int m0 = blockIdx.y * 128, n0 = blockIdx.x * 64;
  const unsigned int as_lds = lds_off(&As[0][0]);

  v8f acc[4];
#pragma unroll
  for (int i = 0; i < 4; ++i) acc[i] = zero8();

  for (int k0 = 0; k0 < EE; k0 += 32) {
    if (wave == 0) {  // A tile: 128 rows x 32 bf16, row stride E elements
      tdm_load_2d_bf16(as_lds, CtxH + (size_t)m0 * EE + k0,
                       /*tile_w=*/32, /*tile_h=*/128,
                       /*row_stride=*/EE, /*tensor_w=*/EE, /*tensor_h=*/MROWS);
    }
#pragma unroll
    for (int i = 0; i < 4; ++i) {  // B tile needs f32->bf16 conversion
      int idx = tid + i * 256, r = idx >> 4, c = idx & 15;
      const float* p = w + (size_t)(n0 + r) * EE + k0 + c * 2;
      Bs[r][c] = pack_bf16(p[0], p[1]);
    }
    __builtin_amdgcn_s_wait_tensorcnt(0);
    __syncthreads();

    v16bf a = load_a_frag(&As[wave * 16 + ln][0], half, 0);
#pragma unroll
    for (int nt = 0; nt < 4; ++nt) {
      v16bf b = load_b_frag(&Bs[nt * 16 + ln][0], half, 0);
      acc[nt] = wmma_bf16(a, b, acc[nt]);
    }
    __syncthreads();
  }

#pragma unroll
  for (int nt = 0; nt < 4; ++nt) {
    int n = n0 + nt * 16 + ln;
    float bv = bias[n];
#pragma unroll
    for (int r = 0; r < 8; ++r) {
      int m = m0 + wave * 16 + r + 8 * half;
      out[(size_t)m * EE + n] = acc[nt][r] + bv;
    }
  }
}

extern "C" void kernel_launch(void* const* d_in, const int* in_sizes, int n_in,
                              void* d_out, int out_size, void* d_ws, size_t ws_size,
                              hipStream_t stream) {
  (void)in_sizes; (void)n_in; (void)out_size; (void)ws_size;
  const float* query     = (const float*)d_in[0];
  const float* attn_bias = (const float*)d_in[1];
  // d_in[2] = attn_mask (causal): reproduced analytically in attn_kernel
  const float* w_in  = (const float*)d_in[3];
  const float* b_in  = (const float*)d_in[4];
  const float* w_out = (const float*)d_in[5];
  const float* b_out = (const float*)d_in[6];
  float* out = (float*)d_out;

  const size_t qkv_elems = (size_t)BB * HH * TT * DD;  // 4M bf16 each
  unsigned short* Qb  = (unsigned short*)d_ws;
  unsigned short* Kb  = Qb + qkv_elems;
  unsigned short* Vb  = Kb + qkv_elems;
  unsigned short* Ctx = Vb + qkv_elems;  // 4M bf16; total ws = 32 MB

  qkv_proj_kernel<<<dim3(N_QKV / 64, MROWS / 128), 256, 0, stream>>>(
      query, w_in, b_in, Qb, Kb, Vb);
  attn_kernel<<<dim3(TT / 128, HH, BB), 256, 0, stream>>>(
      Qb, Kb, Vb, attn_bias, Ctx);
  out_proj_kernel<<<dim3(EE / 64, MROWS / 128), 256, 0, stream>>>(
      (const unsigned short*)Ctx, w_out, b_out, out);
}
